// VSSTokenMambaModule_22282290332047
// MI455X (gfx1250) — matrix operand order repaired and checked by reading, not compiled, not run on hardware
//
#include <hip/hip_runtime.h>
#include <hip/hip_fp16.h>
#include <math.h>

// ---------------- problem constants (from reference) ----------------
#define CB     4      // batch
#define CH     32     // height
#define CW     32     // width
#define CC     192    // channels
#define CDI    384    // inner dim (2.0 * C)
#define CL     1024   // H*W
#define CKD    4      // scan directions
#define CN     16     // SSM state size
#define CR     12     // dt rank
#define CMLP   768    // MLP hidden
#define CROWS  4096   // B * L
#define CDEPTH 9

typedef __attribute__((ext_vector_type(16))) _Float16 v16h;
typedef __attribute__((ext_vector_type(8)))  _Float16 v8h;
typedef __attribute__((ext_vector_type(8)))  float    v8f;

__device__ __forceinline__ float siluf(float x)     { return x / (1.f + __expf(-x)); }
__device__ __forceinline__ float softplusf(float x) { return (x > 20.f) ? x : log1pf(__expf(x)); }
__device__ __forceinline__ float geluf(float x)     { return 0.5f * x * (1.f + erff(x * 0.70710678118654752f)); }

// Scan-step t of direction k reads original flat position perm_idx(k, t).
// k=0: row-major; k=1: transposed (t = w*H + h -> l = h*W + w); k>=2: reversed.
__device__ __forceinline__ int perm_idx(int k, int t) {
  int tt = (k >= 2) ? (CL - 1 - t) : t;
  if (k & 1) { int w = tt >> 5; int h = tt & 31; return h * CW + w; } // H == W == 32
  return tt;
}

// ---------------- LayerNorm over C=192, wave-per-row, f16 output ----------------
__global__ __launch_bounds__(256)
void ln192_kernel(const float* __restrict__ x, const float* __restrict__ g,
                  const float* __restrict__ b, _Float16* __restrict__ out) {
  const int row  = blockIdx.x * 8 + (threadIdx.x >> 5);
  const int lane = threadIdx.x & 31;
  const float* xr = x + (size_t)row * CC;
  float v[6]; float s = 0.f, ss = 0.f;
#pragma unroll
  for (int j = 0; j < 6; ++j) { float t = xr[lane + j * 32]; v[j] = t; s += t; ss += t * t; }
#pragma unroll
  for (int m = 16; m; m >>= 1) { s += __shfl_xor(s, m, 32); ss += __shfl_xor(ss, m, 32); }
  const float mu = s * (1.f / CC);
  const float rs = rsqrtf(ss * (1.f / CC) - mu * mu + 1e-5f);
  _Float16* orow = out + (size_t)row * CC;
#pragma unroll
  for (int j = 0; j < 6; ++j) {
    int e = lane + j * 32;
    orow[e] = (_Float16)((v[j] - mu) * rs * g[e] + b[e]);
  }
}

// ---------------- WMMA f16 GEMM, 256 thr = 8 waves, 128x32 block tile ----------------
#define BM 128
#define BN 32
#define BKC 32

enum { EPI_PLAIN = 0, EPI_SPLIT_SILU = 1, EPI_RESID = 2, EPI_BIAS_GELU = 3, EPI_BIAS_RESID = 4 };

#define CAT16(lo, hi) __builtin_shufflevector(lo, hi, 0,1,2,3,4,5,6,7,8,9,10,11,12,13,14,15)

template<int EPI>
__global__ __launch_bounds__(256)
void gemm_f16w(const _Float16* __restrict__ Ag, const float* __restrict__ Bg,
               int Kdim, int Ndim,
               float* __restrict__ outF, float* __restrict__ outF2,
               _Float16* __restrict__ outH,
               const float* __restrict__ bias, const float* __restrict__ resid) {
  __shared__ __align__(16) _Float16 As[BM][BKC];   // 8 KB, row-major
  __shared__ __align__(16) _Float16 Bs[BN][BKC];   // 2 KB, n-major (transposed on stage-in)
  const int tid  = threadIdx.x;
  const int wave = tid >> 5;
  const int lane = tid & 31;
  const int half = lane >> 4;
  const int ln16 = lane & 15;
  const int row0 = blockIdx.y * BM;
  const int col0 = blockIdx.x * BN;
  v8f acc0 = {}; v8f acc1 = {};

  for (int kk = 0; kk < Kdim; kk += BKC) {
#pragma unroll
    for (int i = 0; i < 2; ++i) {                 // stage A: 128x32 halves via b128
      int s = tid + i * 256;
      int r = s >> 2, seg = (s & 3) * 8;
      *(v8h*)&As[r][seg] = *(const v8h*)(Ag + (size_t)(row0 + r) * Kdim + kk + seg);
    }
#pragma unroll
    for (int i = 0; i < 4; ++i) {                 // stage B: 32x32 f32 -> f16, transposed
      int e = tid + i * 256;
      int n = e & 31, kr = e >> 5;
      Bs[n][kr] = (_Float16)Bg[(size_t)(kk + kr) * Ndim + col0 + n];
    }
    __syncthreads();
    if (kk + BKC < Kdim)                          // global_prefetch_b8 for next A chunk
      __builtin_prefetch(Ag + (size_t)(row0 + (tid >> 1)) * Kdim + kk + BKC, 0, 1);

    // A 16x32 layout: K = 16*half + 0..15 (contiguous, matches B)
    const int m = wave * 16 + ln16;
    v8h alo = *(const v8h*)&As[m][half * 16];
    v8h ahi = *(const v8h*)&As[m][half * 16 + 8];
    v16h afrag = CAT16(alo, ahi);
    // B 32x16 fragment (order shared with A): n = lane&15, K = 16*half + 0..15 (contiguous in Bs[n][*])
    v8h b0lo = *(const v8h*)&Bs[ln16][half * 16];
    v8h b0hi = *(const v8h*)&Bs[ln16][half * 16 + 8];
    v16h bf0 = CAT16(b0lo, b0hi);
    v8h b1lo = *(const v8h*)&Bs[16 + ln16][half * 16];
    v8h b1hi = *(const v8h*)&Bs[16 + ln16][half * 16 + 8];
    v16h bf1 = CAT16(b1lo, b1hi);

    acc0 = __builtin_amdgcn_wmma_f32_16x16x32_f16(false, afrag, false, bf0, (short)0, acc0, false, false);
    acc1 = __builtin_amdgcn_wmma_f32_16x16x32_f16(false, afrag, false, bf1, (short)0, acc1, false, false);
    __syncthreads();
  }

#pragma unroll
  for (int t = 0; t < 2; ++t) {
    v8f acc = t ? acc1 : acc0;
    const int col = col0 + t * 16 + ln16;
#pragma unroll
    for (int i = 0; i < 8; ++i) {
      const int row = row0 + wave * 16 + i + 8 * half;   // C/D layout: M = i + 8*half
      const float v = acc[i];
      const size_t idx = (size_t)row * Ndim + col;
      if (EPI == EPI_PLAIN) {
        outF[idx] = v;
      } else if (EPI == EPI_SPLIT_SILU) {
        if (col < CDI) outF[(size_t)row * CDI + col] = v;
        else           outF2[(size_t)row * CDI + (col - CDI)] = siluf(v);
      } else if (EPI == EPI_RESID) {
        outF[idx] = v + resid[idx];
      } else if (EPI == EPI_BIAS_GELU) {
        outH[idx] = (_Float16)geluf(v + bias[col]);
      } else if (EPI == EPI_BIAS_RESID) {
        outF[idx] = v + bias[col] + resid[idx];
      }
    }
  }
}

// ---------------- direction-projection GEMM: per (b,k) [1024x384]·[384x44->48] ----------------
__global__ __launch_bounds__(256)
void gemm_xdbl(const _Float16* __restrict__ xs_h, const float* __restrict__ xproj,
               float* __restrict__ xdbl) {
  __shared__ __align__(16) _Float16 As[BM][BKC];
  __shared__ __align__(16) _Float16 Bs[BN][BKC];
  const int tid  = threadIdx.x;
  const int wave = tid >> 5;
  const int lane = tid & 31;
  const int half = lane >> 4;
  const int ln16 = lane & 15;
  const int bk   = blockIdx.z;          // b*K + k
  const int kdir = bk & 3;
  const int row0 = blockIdx.y * BM;
  const int col0 = blockIdx.x * BN;
  const _Float16* Ag = xs_h + (size_t)bk * CL * CDI;
  const float*    Bg = xproj + (size_t)kdir * 44 * CDI;   // [c][d] -> transposed load
  float*        outp = xdbl + (size_t)bk * CL * 48;
  v8f acc0 = {}; v8f acc1 = {};

  for (int kk = 0; kk < CDI; kk += BKC) {
#pragma unroll
    for (int i = 0; i < 2; ++i) {
      int s = tid + i * 256;
      int r = s >> 2, seg = (s & 3) * 8;
      *(v8h*)&As[r][seg] = *(const v8h*)(Ag + (size_t)(row0 + r) * CDI + kk + seg);
    }
#pragma unroll
    for (int i = 0; i < 4; ++i) {
      int e = tid + i * 256;
      int n = e & 31, kr = e >> 5;
      int c = col0 + n;
      Bs[n][kr] = (c < 44) ? (_Float16)Bg[(size_t)c * CDI + kk + kr] : (_Float16)0.f;
    }
    __syncthreads();
    const int m = wave * 16 + ln16;
    v8h alo = *(const v8h*)&As[m][half * 16];
    v8h ahi = *(const v8h*)&As[m][half * 16 + 8];
    v16h afrag = CAT16(alo, ahi);
    v8h b0lo = *(const v8h*)&Bs[ln16][half * 16];
    v8h b0hi = *(const v8h*)&Bs[ln16][half * 16 + 8];
    v16h bf0 = CAT16(b0lo, b0hi);
    v8h b1lo = *(const v8h*)&Bs[16 + ln16][half * 16];
    v8h b1hi = *(const v8h*)&Bs[16 + ln16][half * 16 + 8];
    v16h bf1 = CAT16(b1lo, b1hi);
    acc0 = __builtin_amdgcn_wmma_f32_16x16x32_f16(false, afrag, false, bf0, (short)0, acc0, false, false);
    acc1 = __builtin_amdgcn_wmma_f32_16x16x32_f16(false, afrag, false, bf1, (short)0, acc1, false, false);
    __syncthreads();
  }
#pragma unroll
  for (int t = 0; t < 2; ++t) {
    v8f acc = t ? acc1 : acc0;
    const int col = col0 + t * 16 + ln16;
    if (col < 48) {
#pragma unroll
      for (int i = 0; i < 8; ++i) {
        const int row = row0 + wave * 16 + i + 8 * half;
        outp[(size_t)row * 48 + col] = acc[i];
      }
    }
  }
}

// ---------------- depthwise 3x3 conv + bias + SiLU; [B,L,DI] -> [B,L,DI] ----------------
__global__ __launch_bounds__(256)
void dwconv_kernel(const float* __restrict__ xv, const float* __restrict__ cw,
                   const float* __restrict__ cb, float* __restrict__ xct) {
  const int idx = blockIdx.x * blockDim.x + threadIdx.x;
  if (idx >= CB * CL * CDI) return;
  const int d = idx % CDI;
  const int l = (idx / CDI) % CL;
  const int b = idx / (CDI * CL);
  const int h = l >> 5, w = l & 31;
  float s = cb[d];
#pragma unroll
  for (int kh = 0; kh < 3; ++kh) {
    const int hh = h + kh - 1;
    if (hh < 0 || hh >= CH) continue;
#pragma unroll
    for (int kw = 0; kw < 3; ++kw) {
      const int ww = w + kw - 1;
      if (ww < 0 || ww >= CW) continue;
      s += cw[d * 9 + kh * 3 + kw] * xv[((size_t)b * CL + hh * CW + ww) * CDI + d];
    }
  }
  xct[idx] = siluf(s);
}

// ---------------- build xs (f16) for all 4 directions, position-major ----------------
__global__ __launch_bounds__(256)
void gather_xs(const float* __restrict__ xct, _Float16* __restrict__ xs_h) {
  const int idx = blockIdx.x * blockDim.x + threadIdx.x;
  if (idx >= CB * CKD * CL * CDI) return;
  const int d = idx % CDI;
  const int t = (idx / CDI) % CL;
  const int k = (idx / (CDI * CL)) % CKD;
  const int b = idx / (CDI * CL * CKD);
  const int l = perm_idx(k, t);
  xs_h[idx] = (_Float16)xct[((size_t)b * CL + l) * CDI + d];
}

// ---------------- selective scan: block per (b,k), lane per channel d ----------------
__global__ __launch_bounds__(384)
void scan_kernel(const float* __restrict__ xdbl, const float* __restrict__ xct,
                 const float* __restrict__ dtw_g, const float* __restrict__ dtb_g,
                 const float* __restrict__ Alog_g, const float* __restrict__ Ds_g,
                 float* __restrict__ ys) {
  __shared__ float rowbuf[48];            // per-step dts[0..11] | B[12..27] | C[28..43]
  const int bk = blockIdx.x;
  const int b  = bk >> 2, k = bk & 3;
  const int d  = threadIdx.x;
  float Arow[CN], h[CN];
#pragma unroll
  for (int n = 0; n < CN; ++n) {
    Arow[n] = -__expf(Alog_g[(size_t)(k * CDI + d) * CN + n]);
    h[n] = 0.f;
  }
  float dtw[CR];
#pragma unroll
  for (int r = 0; r < CR; ++r) dtw[r] = dtw_g[(size_t)(k * CDI + d) * CR + r];
  const float dtb = dtb_g[k * CDI + d];
  const float Dd  = Ds_g[k * CDI + d];
  const float* xdb  = xdbl + (size_t)bk * CL * 48;
  float*       yout = ys   + (size_t)bk * CL * CDI;

  for (int t = 0; t < CL; ++t) {
    if (d < 48) rowbuf[d] = xdb[t * 48 + d];
    __syncthreads();
    float sdt = dtb;
#pragma unroll
    for (int r = 0; r < CR; ++r) sdt += rowbuf[r] * dtw[r];
    const float delta = softplusf(sdt);
    const int   l = perm_idx(k, t);
    const float u = xct[((size_t)b * CL + l) * CDI + d];
    const float du = delta * u;
    float y = Dd * u;
#pragma unroll
    for (int n = 0; n < CN; ++n) {
      h[n] = __expf(delta * Arow[n]) * h[n] + du * rowbuf[12 + n];
      y += h[n] * rowbuf[28 + n];
    }
    yout[(size_t)t * CDI + d] = y;
    __syncthreads();
  }
}

// ---------------- combine 4 directions + LayerNorm(DI) * silu(z) -> f16 ----------------
__global__ __launch_bounds__(256)
void combine_ln_kernel(const float* __restrict__ ys, const float* __restrict__ zs,
                       const float* __restrict__ g, const float* __restrict__ bb,
                       _Float16* __restrict__ out) {
  const int row  = blockIdx.x * 8 + (threadIdx.x >> 5);   // b*L + l
  const int lane = threadIdx.x & 31;
  const int b = row >> 10, l = row & 1023;
  const int t1 = (l & 31) * CH + (l >> 5);                // inverse transpose index
  const float* y0 = ys + ((size_t)(b * CKD + 0) * CL + l) * CDI;
  const float* y1 = ys + ((size_t)(b * CKD + 1) * CL + t1) * CDI;
  const float* y2 = ys + ((size_t)(b * CKD + 2) * CL + (CL - 1 - l)) * CDI;
  const float* y3 = ys + ((size_t)(b * CKD + 3) * CL + (CL - 1 - t1)) * CDI;
  float v[12]; float s = 0.f, ss = 0.f;
#pragma unroll
  for (int j = 0; j < 12; ++j) {
    int e = lane + j * 32;
    float t = y0[e] + y1[e] + y2[e] + y3[e];
    v[j] = t; s += t; ss += t * t;
  }
#pragma unroll
  for (int m = 16; m; m >>= 1) { s += __shfl_xor(s, m, 32); ss += __shfl_xor(ss, m, 32); }
  const float mu = s * (1.f / CDI);
  const float rs = rsqrtf(ss * (1.f / CDI) - mu * mu + 1e-5f);
#pragma unroll
  for (int j = 0; j < 12; ++j) {
    int e = lane + j * 32;
    float o = (v[j] - mu) * rs * g[e] + bb[e];
    o *= zs[(size_t)row * CDI + e];
    out[(size_t)row * CDI + e] = (_Float16)o;
  }
}

// ---------------- host launcher ----------------
extern "C" void kernel_launch(void* const* d_in, const int* in_sizes, int n_in,
                              void* d_out, int out_size, void* d_ws, size_t ws_size,
                              hipStream_t stream) {
  const float* x_in   = (const float*)d_in[0];
  const float* n1g    = (const float*)d_in[1];
  const float* n1b    = (const float*)d_in[2];
  const float* in_w   = (const float*)d_in[3];
  const float* conv_w = (const float*)d_in[4];
  const float* conv_b = (const float*)d_in[5];
  const float* xproj  = (const float*)d_in[6];
  const float* dt_w   = (const float*)d_in[7];
  const float* dt_b   = (const float*)d_in[8];
  const float* A_logs = (const float*)d_in[9];
  const float* Ds     = (const float*)d_in[10];
  const float* ong    = (const float*)d_in[11];
  const float* onb    = (const float*)d_in[12];
  const float* out_w  = (const float*)d_in[13];
  const float* n2g    = (const float*)d_in[14];
  const float* n2b    = (const float*)d_in[15];
  const float* w1     = (const float*)d_in[16];
  const float* b1     = (const float*)d_in[17];
  const float* w2     = (const float*)d_in[18];
  const float* b2     = (const float*)d_in[19];

  // workspace carve (~75 MB, all L2-resident on MI455X's 192 MB L2)
  char* p = (char*)d_ws;
  auto carve = [&](size_t bytes) -> void* {
    void* r = (void*)p; p += (bytes + 255) & ~(size_t)255; return r;
  };
  float*    X     = (float*)carve((size_t)CROWS * CC * 4);
  float*    XMID  = (float*)carve((size_t)CROWS * CC * 4);
  _Float16* xn_h  = (_Float16*)carve((size_t)CROWS * CC * 2);
  float*    xv    = (float*)carve((size_t)CROWS * CDI * 4);
  float*    zs    = (float*)carve((size_t)CROWS * CDI * 4);
  float*    xct   = (float*)carve((size_t)CROWS * CDI * 4);
  _Float16* xs_h  = (_Float16*)carve((size_t)CB * CKD * CL * CDI * 2);
  float*    xdbl  = (float*)carve((size_t)CB * CKD * CL * 48 * 4);
  float*    ys    = (float*)carve((size_t)CB * CKD * CL * CDI * 4);
  _Float16* yln_h = (_Float16*)carve((size_t)CROWS * CDI * 2);
  _Float16* h1_h  = (_Float16*)carve((size_t)CROWS * CMLP * 2);
  (void)ws_size; (void)in_sizes; (void)n_in; (void)out_size;

  hipMemcpyAsync(X, x_in, (size_t)CROWS * CC * 4, hipMemcpyDeviceToDevice, stream);

  for (int layer = 0; layer < CDEPTH; ++layer) {
    const float* n1g_l = n1g + layer * CC;
    const float* n1b_l = n1b + layer * CC;
    const float* inw_l = in_w + (size_t)layer * CC * 2 * CDI;
    const float* cw_l  = conv_w + (size_t)layer * CDI * 9;
    const float* cb_l  = conv_b + (size_t)layer * CDI;
    const float* xp_l  = xproj + (size_t)layer * CKD * 44 * CDI;
    const float* dtw_l = dt_w + (size_t)layer * CKD * CDI * CR;
    const float* dtb_l = dt_b + (size_t)layer * CKD * CDI;
    const float* Al_l  = A_logs + (size_t)layer * CKD * CDI * CN;
    const float* Ds_l  = Ds + (size_t)layer * CKD * CDI;
    const float* ong_l = ong + (size_t)layer * CDI;
    const float* onb_l = onb + (size_t)layer * CDI;
    const float* ow_l  = out_w + (size_t)layer * CDI * CC;
    const float* n2g_l = n2g + layer * CC;
    const float* n2b_l = n2b + layer * CC;
    const float* w1_l  = w1 + (size_t)layer * CC * CMLP;
    const float* b1_l  = b1 + (size_t)layer * CMLP;
    const float* w2_l  = w2 + (size_t)layer * CMLP * CC;
    const float* b2_l  = b2 + (size_t)layer * CC;

    // 1. LN1 -> f16
    ln192_kernel<<<CROWS / 8, 256, 0, stream>>>(X, n1g_l, n1b_l, xn_h);
    // 2. in-proj GEMM [4096,192]x[192,768], split -> xv | silu(z)
    gemm_f16w<EPI_SPLIT_SILU><<<dim3((2 * CDI) / BN, CROWS / BM), 256, 0, stream>>>(
        xn_h, inw_l, CC, 2 * CDI, xv, zs, nullptr, nullptr, nullptr);
    // 3. depthwise conv + SiLU
    dwconv_kernel<<<(CB * CL * CDI + 255) / 256, 256, 0, stream>>>(xv, cw_l, cb_l, xct);
    // 4. build 4-direction f16 copies (GEMM A)
    gather_xs<<<(CB * CKD * CL * CDI + 255) / 256, 256, 0, stream>>>(xct, xs_h);
    // 5. direction projection: 16 batched [1024,384]x[384,44->48]
    gemm_xdbl<<<dim3(48 / BN + 1, CL / BM, CB * CKD), 256, 0, stream>>>(xs_h, xp_l, xdbl);
    // 6. selective scan (delta fused in-register)
    scan_kernel<<<CB * CKD, CDI, 0, stream>>>(xdbl, xct, dtw_l, dtb_l, Al_l, Ds_l, ys);
    // 7. combine directions + LN(DI) * silu(z) -> f16
    combine_ln_kernel<<<CROWS / 8, 256, 0, stream>>>(ys, zs, ong_l, onb_l, yln_h);
    // 8. out-proj GEMM + residual
    gemm_f16w<EPI_RESID><<<dim3(CC / BN, CROWS / BM), 256, 0, stream>>>(
        yln_h, ow_l, CDI, CC, XMID, nullptr, nullptr, nullptr, X);
    // 9. LN2 -> f16
    ln192_kernel<<<CROWS / 8, 256, 0, stream>>>(XMID, n2g_l, n2b_l, xn_h);
    // 10. MLP up + exact GELU -> f16
    gemm_f16w<EPI_BIAS_GELU><<<dim3(CMLP / BN, CROWS / BM), 256, 0, stream>>>(
        xn_h, w1_l, CC, CMLP, nullptr, nullptr, h1_h, b1_l, nullptr);
    // 11. MLP down + bias + residual (last layer writes d_out directly)
    float* xnext = (layer == CDEPTH - 1) ? (float*)d_out : X;
    gemm_f16w<EPI_BIAS_RESID><<<dim3(CC / BN, CROWS / BM), 256, 0, stream>>>(
        h1_h, w2_l, CMLP, CC, xnext, nullptr, nullptr, b2_l, XMID);
  }
}
